// EncoderGNNAtomBond_76020921140238
// MI455X (gfx1250) — matrix operand order repaired and checked by reading, not compiled, not run on hardware
//
#include <hip/hip_runtime.h>
#include <hip/hip_fp16.h>
#include <math.h>

typedef _Float16 h16;
typedef __attribute__((ext_vector_type(4)))  float    v4f;
typedef __attribute__((ext_vector_type(4)))  _Float16 v4h;
typedef __attribute__((ext_vector_type(8)))  _Float16 v8h;
typedef __attribute__((ext_vector_type(16))) _Float16 v16h;
typedef __attribute__((ext_vector_type(8)))  float    v8f;

#define NNODE    10000
#define SI       128
#define VI       32
#define EDGE_DIM 16
#define RBF_DIM  64
#define CUTOFFF  5.0f
#define EPSV     1e-6f
#define RBF_GAMMA 81.92f                       // 0.5*(64/5)^2
#define RBF_STEP (CUTOFFF/(float)(RBF_DIM-1))
#define PI_F     3.14159265358979f
#define ORD_NEGINF 0x007FFFFFu

// ---------------- WMMA fragment helpers (CDNA5 wave32 layouts) ----------------
// A tile 16(M)x32(K) row-major in LDS, stride in halves (multiple of 8 -> 16B rows).
__device__ __forceinline__ v16h frag_a_lds(const h16* A, int lda){
  int lane = threadIdx.x & 31;
  const h16* p = A + (lane & 15)*lda + ((lane >> 4) << 4);
  union { v16h v; v8h h[2]; } u;
  u.h[0] = *(const v8h*)p;          // K = khi+0..7   (ds_load_b128)
  u.h[1] = *(const v8h*)(p + 8);    // K = khi+8..15
  return u.v;
}
// B fragments stored in "fragment order": [tile][lane][16 halves] contiguous.
__device__ __forceinline__ v16h frag_b_frag(const h16* Wf, int nt){
  return *(const v16h*)(Wf + (nt << 9) + ((threadIdx.x & 31) << 4));
}
__device__ __forceinline__ float silu_f(float x){ return x*(1.f/(1.f+__expf(-x))); }
__device__ __forceinline__ float cutoff_fn(float d){
  return (d < CUTOFFF) ? 0.5f*(__cosf(PI_F*d/CUTOFFF)+1.f) : 0.f;
}
__device__ __forceinline__ unsigned f2ord(float f){
  unsigned u = __float_as_uint(f);
  return (u & 0x80000000u) ? ~u : (u | 0x80000000u);
}
__device__ __forceinline__ float ord2f(unsigned u){
  unsigned b = (u & 0x80000000u) ? (u & 0x7FFFFFFFu) : ~u;
  return __uint_as_float(b);
}

// ---------------- small elementwise kernels ----------------
__global__ void copy_f32_k(const float* __restrict__ in, float* __restrict__ out, int n){
  int i = blockIdx.x*256 + threadIdx.x; if (i < n) out[i] = in[i];
}
__global__ void fill_u32_k(unsigned* __restrict__ p, unsigned v, int n){
  int i = blockIdx.x*256 + threadIdx.x; if (i < n) p[i] = v;
}
// transpose + convert + pad: out[n*KP + k] = (k<K) ? in[k*N + n] : 0   (out: N x KP f16)
__global__ void transpose_cvt_k(const float* __restrict__ in, h16* __restrict__ out,
                                int K, int N, int KP){
  int i = blockIdx.x*256 + threadIdx.x;
  if (i < N*KP){
    int n = i / KP, k = i - n*KP;
    out[i] = (k < K) ? (h16)in[(size_t)k*N + n] : (h16)0.f;
  }
}

// ---------------- layer norm (one node per block, 128 threads) ----------------
__global__ void layernorm_k(const float* __restrict__ s_in, const float* __restrict__ v_in,
                            const float* __restrict__ w, const float* __restrict__ b,
                            float* __restrict__ s_out, float* __restrict__ v_out, int Nn){
  int n = blockIdx.x; if (n >= Nn) return;
  int t = threadIdx.x;
  __shared__ float red[128];
  float x = s_in[(size_t)n*SI + t];
  red[t] = x; __syncthreads();
  for (int o = 64; o; o >>= 1){ if (t < o) red[t] += red[t+o]; __syncthreads(); }
  float mu = red[0] * (1.f/SI); __syncthreads();
  float dx = x - mu;
  red[t] = dx*dx; __syncthreads();
  for (int o = 64; o; o >>= 1){ if (t < o) red[t] += red[t+o]; __syncthreads(); }
  float var = red[0] * (1.f/SI); __syncthreads();
  s_out[(size_t)n*SI + t] = dx * __frsqrt_rn(var + EPSV) * w[t] + b[t];
  float vsq = 0.f, vval = 0.f;
  if (t < 3*VI){ vval = v_in[(size_t)n*3*VI + t]; vsq = vval*vval; }
  red[t] = vsq; __syncthreads();
  for (int o = 64; o; o >>= 1){ if (t < o) red[t] += red[t+o]; __syncthreads(); }
  float vn = __fsqrt_rn(fmaxf(red[0] * (1.f/VI), EPSV));
  if (t < 3*VI) v_out[(size_t)n*3*VI + t] = vval / vn;
}

// ---------------- templated WMMA GEMM: Out = act(A @ W + bias) ----------------
// A: MxKD f32, Wt: transposed f16 weights [ND][KD] (KD multiple of 32), Out: MxND f32.
template<int KD, int ND, int ACT>
__global__ void wmma_gemm_t(const float* __restrict__ A, const h16* __restrict__ Wt,
                            const float* __restrict__ bias, float* __restrict__ Out, int M){
  constexpr int NT  = ND / 16;
  constexpr int NKC = KD / 32;
  __shared__ alignas(16) h16 sA[128*40];
  __shared__ alignas(32) h16 sWf[NT*512];
  int tid = threadIdx.x;
  int lane = tid & 31, wv = tid >> 5;
  int m0 = blockIdx.x * 128;
  v8f acc[NT]; v8f z = {};
#pragma unroll
  for (int i = 0; i < NT; ++i) acc[i] = z;
#pragma unroll 1
  for (int kc = 0; kc < NKC; ++kc){
    int k0 = kc << 5;
    // stage A: 128 rows x 32 k, vectorized 4 floats -> 4 halves
    for (int t = tid; t < 128*8; t += 256){
      int r = t >> 3, g = (t & 7) << 2;
      int gm = m0 + r;
      v4h h4 = {};
      if (gm < M){
        v4f f4 = *(const v4f*)&A[(size_t)gm*KD + k0 + g];
        h4[0] = (h16)f4.x; h4[1] = (h16)f4.y; h4[2] = (h16)f4.z; h4[3] = (h16)f4.w;
      }
      *(v4h*)&sA[r*40 + g] = h4;
    }
    // stage W fragments: lane-contiguous 32B copies
    for (int t = tid; t < NT*32; t += 256){
      int l = t & 31;
      int n = ((t >> 5) << 4) + (l & 15);
      int kb = k0 + ((l >> 4) << 4);
      *(v16h*)&sWf[t << 4] = *(const v16h*)&Wt[(size_t)n*KD + kb];
    }
    if (kc + 1 < NKC) __builtin_prefetch(&Wt[(size_t)(tid & 63)*KD + k0 + 32], 0, 1);
    __syncthreads();
    v16h fa = frag_a_lds(&sA[(wv*16)*40], 40);
#pragma unroll
    for (int nt = 0; nt < NT; ++nt){
      v16h fb = frag_b_frag(sWf, nt);
      acc[nt] = __builtin_amdgcn_wmma_f32_16x16x32_f16(false, fa, false, fb,
                                                       (short)0, acc[nt], false, false);
    }
    __syncthreads();
  }
  int nlo = lane & 15, mhi = (lane >> 4) << 3;
#pragma unroll
  for (int nt = 0; nt < NT; ++nt){
    int n = nt*16 + nlo;
    float bv = bias ? bias[n] : 0.f;
#pragma unroll
    for (int g = 0; g < 8; ++g){
      int m = m0 + wv*16 + mhi + g;
      if (m < M){
        float val = acc[nt][g] + bv;
        if (ACT) val = silu_f(val);
        Out[(size_t)m*ND + n] = val;
      }
    }
  }
}

// ---------------- templated fused edge MLP ----------------
// edge_in built on the fly; 2 back-to-back WMMA GEMMs. 128 threads, 64 edges / block.
// Wt0: [SI][KP] transposed+padded f16; Wt1: [N2][SI] transposed f16.
template<int KIN, int N2>
__global__ void edge_mlp_t(const float* __restrict__ s_ln,
                           const int* __restrict__ srcI, const int* __restrict__ dstI,
                           const float* __restrict__ dArr, const float* __restrict__ efeat,
                           const h16* __restrict__ Wt0, const float* __restrict__ b0,
                           const h16* __restrict__ Wt1, const float* __restrict__ b1,
                           float* __restrict__ aij, float* __restrict__ vijOut, int E){
  constexpr int  KP      = (KIN + 31) & ~31;
  constexpr int  NKC     = KP / 32;
  constexpr int  NT2     = N2 / 16;
  constexpr int  VW      = N2 - SI;
  constexpr bool USE_RBF = (KIN == 320);
  __shared__ int   sSrc[64]; __shared__ int sDst[64]; __shared__ float sD[64];
  __shared__ alignas(16) h16 sA[64*40];
  __shared__ alignas(32) h16 sWf[NT2*512];
  __shared__ alignas(16) h16 sH[64*136];
  int tid = threadIdx.x;
  int lane = tid & 31, wv = tid >> 5;
  int e0 = blockIdx.x * 64;
  if (tid < 64){
    int e = e0 + tid;
    sSrc[tid] = (e < E) ? srcI[e] : 0;
    sDst[tid] = (e < E) ? dstI[e] : 0;
    sD[tid]   = (e < E) ? dArr[e] : 0.f;
  }
  __syncthreads();
  // ---- GEMM1: h = silu(edge_in @ W0 + b0), Nout = SI (8 tiles) ----
  v8f acc1[8]; v8f z = {};
#pragma unroll
  for (int i = 0; i < 8; ++i) acc1[i] = z;
#pragma unroll 1
  for (int kc = 0; kc < NKC; ++kc){
    int k0 = kc << 5;
    for (int idx = tid; idx < 64*32; idx += 128){
      int r = idx >> 5, kk = idx & 31;
      int e = e0 + r, k = k0 + kk;
      float val = 0.f;
      if (e < E && k < KIN){
        if (k < SI)        val = s_ln[(size_t)sDst[r]*SI + k];
        else if (k < 2*SI) val = s_ln[(size_t)sSrc[r]*SI + (k - SI)];
        else {
          int k2 = k - 2*SI;
          if (USE_RBF){ float dd = sD[r] - (float)k2*RBF_STEP; val = __expf(-RBF_GAMMA*dd*dd); }
          else        { val = (k2 == 0) ? sD[r] : efeat[(size_t)e*EDGE_DIM + (k2-1)]; }
        }
      }
      sA[r*40 + kk] = (h16)val;
    }
    for (int t = tid; t < 8*32; t += 128){
      int l = t & 31;
      int n = ((t >> 5) << 4) + (l & 15);
      int kb = k0 + ((l >> 4) << 4);
      *(v16h*)&sWf[t << 4] = *(const v16h*)&Wt0[(size_t)n*KP + kb];
    }
    __syncthreads();
    v16h fa = frag_a_lds(&sA[(wv*16)*40], 40);
#pragma unroll
    for (int nt = 0; nt < 8; ++nt){
      v16h fb = frag_b_frag(sWf, nt);
      acc1[nt] = __builtin_amdgcn_wmma_f32_16x16x32_f16(false, fa, false, fb,
                                                        (short)0, acc1[nt], false, false);
    }
    __syncthreads();
  }
  int nlo = lane & 15, mhi = (lane >> 4) << 3;
#pragma unroll
  for (int nt = 0; nt < 8; ++nt){
    int n = nt*16 + nlo;
    float bb = b0[n];
#pragma unroll
    for (int g = 0; g < 8; ++g){
      float val = silu_f(acc1[nt][g] + bb);
      sH[(wv*16 + mhi + g)*136 + n] = (h16)val;
    }
  }
  __syncthreads();
  // ---- GEMM2: out = h @ W1 + b1, K2 = SI = 128 (4 chunks), Nout = N2 ----
  v8f acc2[NT2];
#pragma unroll
  for (int i = 0; i < NT2; ++i) acc2[i] = z;
#pragma unroll 1
  for (int kc = 0; kc < 4; ++kc){
    int k0 = kc << 5;
    for (int t = tid; t < NT2*32; t += 128){
      int l = t & 31;
      int n = ((t >> 5) << 4) + (l & 15);
      int kb = k0 + ((l >> 4) << 4);
      *(v16h*)&sWf[t << 4] = *(const v16h*)&Wt1[(size_t)n*SI + kb];
    }
    __syncthreads();
    v16h fa = frag_a_lds(&sH[(wv*16)*136 + k0], 136);
#pragma unroll
    for (int nt = 0; nt < NT2; ++nt){
      v16h fb = frag_b_frag(sWf, nt);
      acc2[nt] = __builtin_amdgcn_wmma_f32_16x16x32_f16(false, fa, false, fb,
                                                        (short)0, acc2[nt], false, false);
    }
    __syncthreads();
  }
#pragma unroll
  for (int nt = 0; nt < NT2; ++nt){
    int n = nt*16 + nlo;
    float bb = b1[n];
#pragma unroll
    for (int g = 0; g < 8; ++g){
      int e = e0 + wv*16 + mhi + g;
      if (e < E){
        float val = acc2[nt][g] + bb;
        if (n < SI) aij[(size_t)e*SI + n] = val;
        else        vijOut[(size_t)e*VW + (n - SI)] = val;
      }
    }
  }
}

// ---------------- scatter softmax + aggregation ----------------
__global__ void softmax_max_k(const float* __restrict__ logits, const int* __restrict__ dstI,
                              unsigned* __restrict__ nmax, int E){
  int i = blockIdx.x*256 + threadIdx.x;
  if (i < E*SI){
    int e = i >> 7, c = i & 127;
    atomicMax(&nmax[(size_t)dstI[e]*SI + c], f2ord(logits[i]));
  }
}
__global__ void softmax_expsum_k(float* __restrict__ logits, const int* __restrict__ dstI,
                                 const unsigned* __restrict__ nmax, float* __restrict__ den, int E){
  int i = blockIdx.x*256 + threadIdx.x;
  if (i < E*SI){
    int e = i >> 7, c = i & 127;
    float m = ord2f(nmax[(size_t)dstI[e]*SI + c]);
    if (!isfinite(m)) m = 0.f;
    float ex = __expf(logits[i] - m);
    logits[i] = ex;
    atomicAdd(&den[(size_t)dstI[e]*SI + c], ex);
  }
}
__global__ void agg_scalar_k(const float* __restrict__ ex, const float* __restrict__ den,
                             const int* __restrict__ dstI, const int* __restrict__ srcI,
                             const float* __restrict__ sb, const float* __restrict__ dArr,
                             float* __restrict__ s_agg, int E, int use_rbf){
  int e = blockIdx.x; if (e >= E) return;
  int c = threadIdx.x;
  int dn = dstI[e], sn = srcI[e];
  float w = ex[(size_t)e*SI + c] / (den[(size_t)dn*SI + c] + 1e-16f);
  float ns = w * sb[(size_t)sn*SI + c];
  if (use_rbf) ns *= cutoff_fn(dArr[e]);
  atomicAdd(&s_agg[(size_t)dn*SI + c], ns);
}
__global__ void agg_vector_k(const float* __restrict__ vij, const float* __restrict__ avec,
                             const int* __restrict__ dstI, const int* __restrict__ srcI,
                             const float* __restrict__ vb, const float* __restrict__ dArr,
                             float* __restrict__ v_agg, int E, int vw, int has_v_in, int use_rbf){
  int e = blockIdx.x; if (e >= E) return;
  int j = threadIdx.x;              // 0..95: comp*32 + c
  int comp = j >> 5, cc = j & 31;
  float nv = avec[(size_t)e*3 + comp] * vij[(size_t)e*vw + cc];
  if (has_v_in) nv += vij[(size_t)e*vw + VI + cc] * vb[(size_t)srcI[e]*(3*VI) + j];
  if (use_rbf) nv *= cutoff_fn(dArr[e]);
  atomicAdd(&v_agg[(size_t)dstI[e]*(3*VI) + j], nv);
}
__global__ void agg_count_k(const int* __restrict__ dstI, float* __restrict__ cnt, int E){
  int i = blockIdx.x*256 + threadIdx.x;
  if (i < E) atomicAdd(&cnt[dstI[i]], 1.f);
}
__global__ void node_update_k(const float* __restrict__ s_ln, const float* __restrict__ s_agg,
                              const float* __restrict__ v_ln, const float* __restrict__ v_agg,
                              const float* __restrict__ cnt,
                              float* __restrict__ s_out, float* __restrict__ v_out, int Nn){
  int i = blockIdx.x*256 + threadIdx.x;
  int total_s = Nn*SI;
  if (i < total_s){
    s_out[i] = s_ln[i] + s_agg[i];
  } else {
    int j = i - total_s;
    if (j < Nn*3*VI){
      int n = j / (3*VI);
      v_out[j] = v_ln[j] + v_agg[j] / fmaxf(cnt[n], 1.f);
    }
  }
}

// ---------------- gated equivariant block glue ----------------
__global__ void build_sc_k(const float* __restrict__ s, const float* __restrict__ vv,
                           float* __restrict__ sc, int Nn){
  int i = blockIdx.x*256 + threadIdx.x;
  if (i < Nn*(SI+VI)){
    int n = i / (SI+VI), c = i - n*(SI+VI);
    if (c < SI) sc[i] = s[(size_t)n*SI + c];
    else {
      int cc = c - SI;
      float sum = 0.f;
#pragma unroll
      for (int comp = 0; comp < 3; ++comp){
        float x = vv[((size_t)n*3 + comp)*(2*VI) + cc];
        sum += x*x;
      }
      sc[i] = __fsqrt_rn(fmaxf(sum, EPSV));
    }
  }
}
__global__ void gate_apply_k(const float* __restrict__ sc2, const float* __restrict__ vv,
                             float* __restrict__ vout, int Nn){
  int n = blockIdx.x; if (n >= Nn) return;
  int j = threadIdx.x;              // 0..95
  int comp = j >> 5, cc = j & 31;
  float gate = sc2[(size_t)n*(VI+SI) + cc];
  vout[(size_t)n*(3*VI) + j] = gate * vv[((size_t)n*3 + comp)*(2*VI) + VI + cc];
}
__global__ void residual_s_k(float* __restrict__ s, const float* __restrict__ sc2, int Nn){
  int i = blockIdx.x*256 + threadIdx.x;
  if (i < Nn*SI){
    int n = i >> 7, c = i & 127;
    s[i] += sc2[(size_t)n*(VI+SI) + VI + c];
  }
}
__global__ void residual_v_k(float* __restrict__ v, const float* __restrict__ dv, int n){
  int i = blockIdx.x*256 + threadIdx.x; if (i < n) v[i] += dv[i];
}

// =======================================================================
extern "C" void kernel_launch(void* const* d_in, const int* in_sizes, int n_in,
                              void* d_out, int out_size, void* d_ws, size_t ws_size,
                              hipStream_t stream){
  (void)in_sizes; (void)out_size; (void)ws_size;
  const int N  = NNODE;
  const int EL = 160000, EG = 390000;

  const float* s_in = (const float*)d_in[0];
  const float* v_in = (const float*)d_in[1];
  const int*   ei_l = (const int*)d_in[2];
  const float* d_l  = (const float*)d_in[3];
  const float* a_l  = (const float*)d_in[4];
  const int*   ei_g = (const int*)d_in[5];
  const float* d_g  = (const float*)d_in[6];
  const float* a_g  = (const float*)d_in[7];
  const float* e_g  = (const float*)d_in[8];
  // d_in[9] = batch (unused by reference output)

  // params: jax pytree flatten order (dict keys sorted; ASCII 'W'<'b'<'e'<'l'<'s'<'v')
  const float* P[96] = {nullptr};
  for (int i = 0; i + 10 < n_in && i < 96; ++i) P[i] = (const float*)d_in[10 + i];
  const float *Ws0[5]={0},*Ws1[5]={0},*Wv0[5]={0},*Wv1[5]={0},*bs0[5]={0},*bs1[5]={0},
              *eW0[5]={0},*eW1[5]={0},*eb0[5]={0},*eb1[5]={0},*lnb[5]={0},*lnw[5]={0},
              *scW[5]={0},*scB[5]={0},*vecW[5]={0},*WsF[5]={0},*bsF[5]={0};
  {
    int c = 0;
    for (int i = 0; i < 5; ++i){
      bool mlp = (i < 4);
      if (mlp){ Ws0[i]=P[c++]; Ws1[i]=P[c++]; Wv0[i]=P[c++]; Wv1[i]=P[c++]; bs0[i]=P[c++]; bs1[i]=P[c++]; }
      else    { WsF[i]=P[c++]; Wv0[i]=P[c++]; bsF[i]=P[c++]; }
      eW0[i]=P[c++]; eW1[i]=P[c++]; eb0[i]=P[c++]; eb1[i]=P[c++];
      lnb[i]=P[c++]; lnw[i]=P[c++]; scW[i]=P[c++]; scB[i]=P[c++];
      if (i > 0) vecW[i]=P[c++];
    }
  }

  // ---- workspace bump allocator ----
  char* wp = (char*)d_ws;
  auto alloc = [&](size_t bytes)->void*{ void* p = wp; wp += (bytes + 255) & ~(size_t)255; return p; };
  float* s_ln  = (float*)alloc((size_t)N*SI*4);
  float* v_ln  = (float*)alloc((size_t)N*3*VI*4);
  float* sb    = (float*)alloc((size_t)N*SI*4);
  float* vbuf  = (float*)alloc((size_t)N*3*VI*4);
  float* aij   = (float*)alloc((size_t)EG*SI*4);
  float* vij   = (float*)alloc((size_t)EG*64*4);
  unsigned* nmax = (unsigned*)alloc((size_t)N*SI*4);
  float* den   = (float*)alloc((size_t)N*SI*4);
  float* s_agg = (float*)alloc((size_t)N*SI*4);
  float* v_agg = (float*)alloc((size_t)N*3*VI*4);
  float* cnt   = (float*)alloc((size_t)N*4);
  float* vvb   = (float*)alloc((size_t)N*3*(2*VI)*4);
  float* scb   = (float*)alloc((size_t)N*(SI+VI)*4);
  float* hbuf  = (float*)alloc((size_t)N*(SI+VI)*4);
  float* sc2   = (float*)alloc((size_t)N*(SI+VI)*4);
  float* vout3 = (float*)alloc((size_t)N*3*VI*4);
  // transposed f16 weight slots ([N][K_padded])
  h16* w_e0  = (h16*)alloc((size_t)SI*320*2);          // [128][288 or 320]
  h16* w_e1  = (h16*)alloc((size_t)192*SI*2);          // [N2][128]
  h16* w_sc  = (h16*)alloc((size_t)SI*SI*2);           // [128][128]
  h16* w_vec = (h16*)alloc((size_t)VI*VI*2);           // [32][32]
  h16* w_wv0 = (h16*)alloc((size_t)(2*VI)*VI*2);       // [64][32]
  h16* w_ws0 = (h16*)alloc((size_t)SI*(SI+VI)*2);      // [128][160]
  h16* w_ws1 = (h16*)alloc((size_t)(SI+VI)*SI*2);      // [160][128]
  h16* w_wv1 = (h16*)alloc((size_t)VI*VI*2);           // [32][32]
  h16* w_wsF = (h16*)alloc((size_t)(SI+VI)*(SI+VI)*2); // [160][160]

  float* s_cur = (float*)d_out;          // N x 128
  float* v_cur = s_cur + (size_t)N*SI;   // N x 3 x 32

  auto cvtT = [&](const float* src, h16* dst, int K, int Nc, int KP){
    if (src) transpose_cvt_k<<<(Nc*KP+255)/256, 256, 0, stream>>>(src, dst, K, Nc, KP);
  };

  copy_f32_k<<<(N*SI+255)/256, 256, 0, stream>>>(s_in, s_cur, N*SI);
  copy_f32_k<<<(N*3*VI+255)/256, 256, 0, stream>>>(v_in, v_cur, N*3*VI);

  const int e_in_tab[5] = {273, 320, 320, 273, 320};
  const int n2_tab[5]   = {160, 192, 192, 192, 192};
  const int isglob[5]   = {1, 0, 0, 1, 0};

  for (int L = 0; L < 5; ++L){
    const int has_v = (L > 0), mlp = (L < 4), glob = isglob[L];
    const int Kin = e_in_tab[L], N2 = n2_tab[L], vw = N2 - SI;
    const int KPe = (Kin + 31) & ~31;
    const int E = glob ? EG : EL;
    const int* srcI = glob ? ei_g : ei_l;
    const int* dstI = srcI + E;
    const float* dA = glob ? d_g : d_l;
    const float* aA = glob ? a_g : a_l;
    const float* ef = glob ? e_g : nullptr;

    // convert + transpose weights to f16
    cvtT(eW0[L], w_e0, Kin, SI, KPe);
    cvtT(eW1[L], w_e1, SI, N2, SI);
    cvtT(scW[L], w_sc, SI, SI, SI);
    if (has_v) cvtT(vecW[L], w_vec, VI, VI, VI);
    cvtT(Wv0[L], w_wv0, VI, 2*VI, VI);
    if (mlp){
      cvtT(Ws0[L], w_ws0, SI+VI, SI, SI+VI);
      cvtT(Ws1[L], w_ws1, SI, SI+VI, SI);
      cvtT(Wv1[L], w_wv1, VI, VI, VI);
    } else {
      cvtT(WsF[L], w_wsF, SI+VI, SI+VI, SI+VI);
    }

    // layer norm
    layernorm_k<<<N, 128, 0, stream>>>(s_cur, v_cur, lnw[L], lnb[L], s_ln, v_ln, N);

    // sb = s_ln @ scalar_W + scalar_b ; vb = v_ln @ vector_W (if has_v)
    wmma_gemm_t<SI,SI,0><<<(N+127)/128, 256, 0, stream>>>(s_ln, w_sc, scB[L], sb, N);
    const float* vb = v_ln;
    if (has_v){
      wmma_gemm_t<VI,VI,0><<<(3*N+127)/128, 256, 0, stream>>>(v_ln, w_vec, nullptr, vbuf, 3*N);
      vb = vbuf;
    }

    // fused edge MLP (WMMA)
    int eblocks = (E + 63) / 64;
    if (glob && !has_v)
      edge_mlp_t<273,160><<<eblocks, 128, 0, stream>>>(s_ln, srcI, dstI, dA, ef,
                                                       w_e0, eb0[L], w_e1, eb1[L], aij, vij, E);
    else if (glob)
      edge_mlp_t<273,192><<<eblocks, 128, 0, stream>>>(s_ln, srcI, dstI, dA, ef,
                                                       w_e0, eb0[L], w_e1, eb1[L], aij, vij, E);
    else
      edge_mlp_t<320,192><<<eblocks, 128, 0, stream>>>(s_ln, srcI, dstI, dA, ef,
                                                       w_e0, eb0[L], w_e1, eb1[L], aij, vij, E);

    // scatter softmax over dst + aggregation
    fill_u32_k<<<(N*SI+255)/256, 256, 0, stream>>>(nmax, ORD_NEGINF, N*SI);
    fill_u32_k<<<(N*SI+255)/256, 256, 0, stream>>>((unsigned*)den, 0u, N*SI);
    fill_u32_k<<<(N*SI+255)/256, 256, 0, stream>>>((unsigned*)s_agg, 0u, N*SI);
    fill_u32_k<<<(N*3*VI+255)/256, 256, 0, stream>>>((unsigned*)v_agg, 0u, N*3*VI);
    fill_u32_k<<<(N+255)/256, 256, 0, stream>>>((unsigned*)cnt, 0u, N);
    softmax_max_k<<<(E*SI+255)/256, 256, 0, stream>>>(aij, dstI, nmax, E);
    softmax_expsum_k<<<(E*SI+255)/256, 256, 0, stream>>>(aij, dstI, nmax, den, E);
    agg_scalar_k<<<E, 128, 0, stream>>>(aij, den, dstI, srcI, sb, dA, s_agg, E, !glob);
    agg_vector_k<<<E, 96, 0, stream>>>(vij, aA, dstI, srcI, vb, dA, v_agg, E, vw, has_v, !glob);
    agg_count_k<<<(E+255)/256, 256, 0, stream>>>(dstI, cnt, E);
    node_update_k<<<(N*(SI+3*VI)+255)/256, 256, 0, stream>>>(s_ln, s_agg, v_ln, v_agg, cnt,
                                                             s_cur, v_cur, N);

    // gated equivariant block
    wmma_gemm_t<VI,2*VI,0><<<(3*N+127)/128, 256, 0, stream>>>(v_cur, w_wv0, nullptr, vvb, 3*N);
    build_sc_k<<<(N*(SI+VI)+255)/256, 256, 0, stream>>>(s_cur, vvb, scb, N);
    if (mlp){
      wmma_gemm_t<SI+VI,SI,1><<<(N+127)/128, 256, 0, stream>>>(scb, w_ws0, bs0[L], hbuf, N);
      wmma_gemm_t<SI,SI+VI,0><<<(N+127)/128, 256, 0, stream>>>(hbuf, w_ws1, bs1[L], sc2, N);
    } else {
      wmma_gemm_t<SI+VI,SI+VI,0><<<(N+127)/128, 256, 0, stream>>>(scb, w_wsF, bsF[L], sc2, N);
    }
    gate_apply_k<<<N, 96, 0, stream>>>(sc2, vvb, vout3, N);
    const float* dv = vout3;
    if (mlp){
      wmma_gemm_t<VI,VI,0><<<(3*N+127)/128, 256, 0, stream>>>(vout3, w_wv1, nullptr, hbuf, 3*N);
      dv = hbuf;
    }
    residual_s_k<<<(N*SI+255)/256, 256, 0, stream>>>(s_cur, sc2, N);
    residual_v_k<<<(N*3*VI+255)/256, 256, 0, stream>>>(v_cur, dv, N*3*VI);
  }
}
